// StructureModuleV46_66949950210593
// MI455X (gfx1250) — matrix operand order repaired and checked by reading, not compiled, not run on hardware
//
#include <hip/hip_runtime.h>
#include <math.h>

// ---------------------------------------------------------------------------
// Problem constants (match reference)
// ---------------------------------------------------------------------------
#define NN    1024
#define CC    384
#define HH    12
#define PP    4
#define CHH   32
#define DPP   128
#define HIDD  512
#define EE    32768
#define ECH   4096          // edge chunk
#define HP3   144           // H*P*3
#define HP3P  160           // padded to 32
#define DIN   896           // 2C + DP
#define T4    1536
#define SCALE 0.17677669529663687f   // 32^-0.5
#define LNEPS 1e-5f

typedef unsigned short u16;
typedef __attribute__((ext_vector_type(16))) __bf16 v16bf;
typedef __attribute__((ext_vector_type(8)))  float  v8f;

union BfPack { uint4 u; __bf16 b[8]; };

__device__ inline u16 f2bf(float f) {
  unsigned int u = __float_as_uint(f);
  unsigned int r = u + 0x7FFFu + ((u >> 16) & 1u);
  return (u16)(r >> 16);
}

// ---------------------------------------------------------------------------
// WMMA operand loaders (ISA 7.12.2 layouts, 16-bit 16x16x32)
// A (16x32 MxK): lane<16 -> M=lane, K-half 0; lane>=16 -> M=lane-16, K-half +8
//   VGPR r: K = (r<4 ? 2r : 16+2(r-4)) + lo + 8*laneHi
// B (32x16 KxN), stored here as Bt[N][K]: lane gives N, same K order as A:
//   K = (e<8 ? e : 16+(e-8)) + 8*laneHi
// ---------------------------------------------------------------------------
__device__ inline v16bf load_a(const u16* base, long row, int ldk, int k0, int laneHi) {
  const u16* pp = base + row * (long)ldk + k0 + (laneHi ? 8 : 0);
  BfPack c0, c1;
  c0.u = *(const uint4*)(pp);
  c1.u = *(const uint4*)(pp + 16);
  v16bf r;
#pragma unroll
  for (int i = 0; i < 8; ++i) { r[i] = c0.b[i]; r[8 + i] = c1.b[i]; }
  return r;
}

__device__ inline v16bf load_b(const u16* base, long row, int ldk, int k0, int laneHi) {
  const u16* pp = base + row * (long)ldk + k0 + laneHi * 8;
  BfPack c0, c1;
  c0.u = *(const uint4*)(pp);
  c1.u = *(const uint4*)(pp + 16);
  v16bf r;
#pragma unroll
  for (int i = 0; i < 8; ++i) { r[i] = c0.b[i]; r[8 + i] = c1.b[i]; }
  return r;
}

// ---------------------------------------------------------------------------
// Generic bf16 WMMA GEMM:  C[M,Nc](ldc) = act( A[M,Kp] @ Bt[*,Kp]^T + bias (+C) )
// wave computes a 16x64 strip: 4 accumulators share one A fragment per K-step
// (4x A-load reuse). Wave index is readfirstlane'd so all tile bookkeeping is
// scalar: guards become s_cbranch (no EXEC manipulation around WMMA), and the
// dominant nvalid==4 path is branch-free with 4 independent accumulators
// (no WMMA->WMMA hazard NOPs). act: 0 none, 1 relu, 2 silu
// ---------------------------------------------------------------------------
__global__ void k_gemm(const u16* __restrict__ A, const u16* __restrict__ Bt,
                       const float* __restrict__ bias, float* __restrict__ C,
                       int M, int Nc, int Kp, int ldc, int act, int accum) {
  int wave = __builtin_amdgcn_readfirstlane(
      blockIdx.x * (blockDim.x >> 5) + (threadIdx.x >> 5));
  int lane = threadIdx.x & 31;
  int tilesN  = (Nc + 15) >> 4;
  int tilesN4 = (tilesN + 3) >> 2;
  int tilesM  = M >> 4;
  if (wave >= tilesM * tilesN4) return;
  int tm = wave / tilesN4, tg = wave % tilesN4;
  long m0 = (long)tm * 16;
  int laneHi = lane >> 4;
  int lm = lane & 15;
  long arow = m0 + lm;
  int nvalid = __builtin_amdgcn_readfirstlane(
      tilesN - tg * 4 > 4 ? 4 : tilesN - tg * 4);   // >= 1, scalar
  long br0 = (long)(tg * 4 + 0) * 16 + lm;
  long br1 = br0 + 16, br2 = br0 + 32, br3 = br0 + 48;
  v8f acc0 = {}, acc1 = {}, acc2 = {}, acc3 = {};
  if (nvalid == 4) {
    // fast path: branch-free inner loop, 4 independent accumulators
    for (int k0 = 0; k0 < Kp; k0 += 32) {
      v16bf a  = load_a(A,  arow, Kp, k0, laneHi);
      v16bf b0 = load_b(Bt, br0,  Kp, k0, laneHi);
      v16bf b1 = load_b(Bt, br1,  Kp, k0, laneHi);
      v16bf b2 = load_b(Bt, br2,  Kp, k0, laneHi);
      v16bf b3 = load_b(Bt, br3,  Kp, k0, laneHi);
      acc0 = __builtin_amdgcn_wmma_f32_16x16x32_bf16(false, a, false, b0, (short)0,
                                                     acc0, false, false);
      acc1 = __builtin_amdgcn_wmma_f32_16x16x32_bf16(false, a, false, b1, (short)0,
                                                     acc1, false, false);
      acc2 = __builtin_amdgcn_wmma_f32_16x16x32_bf16(false, a, false, b2, (short)0,
                                                     acc2, false, false);
      acc3 = __builtin_amdgcn_wmma_f32_16x16x32_bf16(false, a, false, b3, (short)0,
                                                     acc3, false, false);
    }
  } else {
    // N-edge path: guards are scalar (nvalid in SGPR) -> s_cbranch, EXEC intact
    for (int k0 = 0; k0 < Kp; k0 += 32) {
      v16bf a  = load_a(A,  arow, Kp, k0, laneHi);
      v16bf b0 = load_b(Bt, br0,  Kp, k0, laneHi);
      acc0 = __builtin_amdgcn_wmma_f32_16x16x32_bf16(false, a, false, b0, (short)0,
                                                     acc0, false, false);
      if (nvalid > 1) {
        v16bf b1 = load_b(Bt, br1, Kp, k0, laneHi);
        acc1 = __builtin_amdgcn_wmma_f32_16x16x32_bf16(false, a, false, b1, (short)0,
                                                       acc1, false, false);
      }
      if (nvalid > 2) {
        v16bf b2 = load_b(Bt, br2, Kp, k0, laneHi);
        acc2 = __builtin_amdgcn_wmma_f32_16x16x32_bf16(false, a, false, b2, (short)0,
                                                       acc2, false, false);
      }
    }
  }
  v8f accs[4] = {acc0, acc1, acc2, acc3};
#pragma unroll
  for (int t = 0; t < 4; ++t) {
    if (t >= nvalid) break;
    int n = (tg * 4 + t) * 16 + lm;
    if (n >= Nc) continue;
#pragma unroll
    for (int r = 0; r < 8; ++r) {
      long m = m0 + r + 8 * laneHi;
      float v = accs[t][r];
      if (bias) v += bias[n];
      long idx = m * (long)ldc + n;
      if (accum) v += C[idx];
      if (act == 1)      v = v > 0.f ? v : 0.f;
      else if (act == 2) v = v / (1.f + __expf(-v));
      C[idx] = v;
    }
  }
}

// ---------------------------------------------------------------------------
// fp32 -> bf16 row-major with K padding
// ---------------------------------------------------------------------------
__global__ void k_cvtA(const float* __restrict__ A, u16* __restrict__ Abf,
                       int M, int K, int Kp) {
  long idx = (long)blockIdx.x * blockDim.x + threadIdx.x;
  if (idx >= (long)M * Kp) return;
  long m = idx / Kp; int k = (int)(idx % Kp);
  Abf[idx] = f2bf(k < K ? A[m * (long)K + k] : 0.f);
}

// fp32 W[K,Nc] -> bf16 Bt[Np,Kp] transposed + zero padded
__global__ void k_cvtWt(const float* __restrict__ W, u16* __restrict__ Bt,
                        int K, int Nc, int Kp, int Np) {
  long idx = (long)blockIdx.x * blockDim.x + threadIdx.x;
  if (idx >= (long)Np * Kp) return;
  int n = (int)(idx / Kp); int k = (int)(idx % Kp);
  float v = (n < Nc && k < K) ? W[(long)k * Nc + n] : 0.f;
  Bt[idx] = f2bf(v);
}

__global__ void k_copy(const float* __restrict__ a, float* __restrict__ b, long n) {
  long i = (long)blockIdx.x * blockDim.x + threadIdx.x;
  if (i < n) b[i] = a[i];
}

// ---------------------------------------------------------------------------
// pair bias: pb[n,m,h] = sum_d pair[n,m,d]*pbw[d,h] + pbb[h]   (wave per (n,m))
// bandwidth-bound: coalesced 128-float row read, shfl_xor reduction (wave32)
// ---------------------------------------------------------------------------
__global__ void k_pair_bias(const float* __restrict__ pair,
                            const float* __restrict__ pbw,
                            const float* __restrict__ pbb,
                            float* __restrict__ pb) {
  long wid = (long)blockIdx.x * (blockDim.x >> 5) + (threadIdx.x >> 5);
  if (wid >= (long)NN * NN) return;
  int lane = threadIdx.x & 31;
  const float* pr = pair + wid * DPP;
  float acc[HH];
#pragma unroll
  for (int h = 0; h < HH; ++h) acc[h] = 0.f;
  for (int d0 = 0; d0 < DPP; d0 += 32) {
    float pv = pr[d0 + lane];
    const float* w = pbw + (d0 + lane) * HH;
#pragma unroll
    for (int h = 0; h < HH; ++h) acc[h] += pv * w[h];
  }
#pragma unroll
  for (int h = 0; h < HH; ++h) {
    float s = acc[h];
    for (int off = 16; off; off >>= 1) s += __shfl_xor(s, off, 32);
    acc[h] = s;
  }
  if (lane < HH) pb[wid * HH + lane] = acc[lane] + pbb[lane];
}

// ---------------------------------------------------------------------------
// rigid transform of points: out[n,hp,:] = pts[n,hp,:] @ rot[n] + trans[n]
// ---------------------------------------------------------------------------
__global__ void k_glob(const float* __restrict__ pts, const float* __restrict__ rot,
                       const float* __restrict__ trans, float* __restrict__ out) {
  int idx = blockIdx.x * blockDim.x + threadIdx.x;
  if (idx >= NN * (HH * PP)) return;
  int n = idx / (HH * PP), hp = idx % (HH * PP);
  const float* p = pts + (long)n * HP3 + hp * 3;
  const float* R = rot + (long)n * 9;
  float x = p[0], y = p[1], z = p[2];
  float* o = out + (long)n * HP3 + hp * 3;
  o[0] = x * R[0] + y * R[3] + z * R[6] + trans[n * 3 + 0];
  o[1] = x * R[1] + y * R[4] + z * R[7] + trans[n * 3 + 1];
  o[2] = x * R[2] + y * R[5] + z * R[8] + trans[n * 3 + 2];
}

// ---------------------------------------------------------------------------
// attention logits for head h (wave per 16x16 tile):
//   L[n,m] = SCALE*(q_h . k_h)  +  pb[n,m,h]  -  0.5*SCALE*|qg-kg|^2   (masked)
// scalar QK^T via one v_wmma per tile (K = CH = 32 exactly)
// ---------------------------------------------------------------------------
__global__ void k_attn_logits(const u16* __restrict__ qbf, const u16* __restrict__ kbf,
                              int k0col, const float* __restrict__ qg,
                              const float* __restrict__ kg, int hp0,
                              const float* __restrict__ pb, int h,
                              const float* __restrict__ mask, float* __restrict__ L) {
  int wave = __builtin_amdgcn_readfirstlane(
      blockIdx.x * (blockDim.x >> 5) + (threadIdx.x >> 5));
  int lane = threadIdx.x & 31;
  const int tiles = (NN / 16) * (NN / 16);
  if (wave >= tiles) return;
  int tm = wave / (NN / 16), tn = wave % (NN / 16);
  long m0 = (long)tm * 16, n0 = (long)tn * 16;
  int laneHi = lane >> 4;
  int lm = lane & 15;
  v8f acc = {};
  v16bf a = load_a(qbf, m0 + lm, CC, k0col, laneHi);   // query rows
  v16bf b = load_b(kbf, n0 + lm, CC, k0col, laneHi);   // key rows (as Bt)
  acc = __builtin_amdgcn_wmma_f32_16x16x32_bf16(false, a, false, b, (short)0, acc,
                                                false, false);
  int mk = (int)n0 + lm;
  const float* kgp = kg + (long)mk * HP3 + hp0;
  float kv[12];
#pragma unroll
  for (int j = 0; j < 12; ++j) kv[j] = kgp[j];
  float maskm = mask[mk];
#pragma unroll
  for (int r = 0; r < 8; ++r) {
    long nq = m0 + r + 8 * laneHi;
    const float* qgp = qg + nq * HP3 + hp0;
    float s = 0.f;
#pragma unroll
    for (int j = 0; j < 12; ++j) { float dd = qgp[j] - kv[j]; s += dd * dd; }
    float val = acc[r] * SCALE - 0.5f * SCALE * s + pb[(nq * NN + mk) * HH + h];
    if (mask[nq] == 0.f || maskm == 0.f) val = -1e9f;
    L[nq * NN + mk] = val;
  }
}

// ---------------------------------------------------------------------------
// block reductions / softmax / layernorm
// ---------------------------------------------------------------------------
__device__ inline float blk_reduce(float v, float* sbuf, int isMax) {
  int lane = threadIdx.x & 31, w = threadIdx.x >> 5;
  for (int off = 16; off; off >>= 1) {
    float o = __shfl_xor(v, off, 32);
    v = isMax ? fmaxf(v, o) : v + o;
  }
  if (lane == 0) sbuf[w] = v;
  __syncthreads();
  int nw = blockDim.x >> 5;
  v = (threadIdx.x < nw) ? sbuf[threadIdx.x] : (isMax ? -3.4e38f : 0.f);
  if (w == 0) {
    for (int off = 16; off; off >>= 1) {
      float o = __shfl_xor(v, off, 32);
      v = isMax ? fmaxf(v, o) : v + o;
    }
    if (lane == 0) sbuf[0] = v;
  }
  __syncthreads();
  v = sbuf[0];
  __syncthreads();
  return v;
}

__global__ void k_softmax(float* __restrict__ L, u16* __restrict__ Abf) {
  __shared__ float sbuf[32];
  long row = blockIdx.x;
  float* r = L + row * NN;
  float mx = -3.4e38f;
  for (int i = threadIdx.x; i < NN; i += blockDim.x) mx = fmaxf(mx, r[i]);
  mx = blk_reduce(mx, sbuf, 1);
  float s = 0.f;
  for (int i = threadIdx.x; i < NN; i += blockDim.x) {
    float e = __expf(r[i] - mx);
    r[i] = e; s += e;
  }
  s = blk_reduce(s, sbuf, 0);
  float inv = 1.f / s;
  for (int i = threadIdx.x; i < NN; i += blockDim.x) {
    float v = r[i] * inv;
    r[i] = v;
    Abf[row * NN + i] = f2bf(v);
  }
}

// out = LN(x (+res)) * g + b   ; out may alias x or res (same-index RMW only)
__global__ void k_add_ln(const float* __restrict__ x, const float* __restrict__ res,
                         const float* __restrict__ g, const float* __restrict__ bb,
                         float* __restrict__ out) {
  __shared__ float sbuf[32];
  long row = blockIdx.x;
  const float* xr = x + row * CC;
  const float* rr = res ? res + row * CC : nullptr;
  float s = 0.f, s2 = 0.f;
  for (int i = threadIdx.x; i < CC; i += blockDim.x) {
    float v = xr[i] + (rr ? rr[i] : 0.f);
    s += v; s2 += v * v;
  }
  s  = blk_reduce(s,  sbuf, 0);
  s2 = blk_reduce(s2, sbuf, 0);
  float mean = s / CC;
  float var  = s2 / CC - mean * mean;
  float inv  = rsqrtf(var + LNEPS);
  for (int i = threadIdx.x; i < CC; i += blockDim.x) {
    float v = xr[i] + (rr ? rr[i] : 0.f);
    out[row * CC + i] = (v - mean) * inv * g[i] + bb[i];
  }
}

// ---------------------------------------------------------------------------
// EGNN helper kernels
// ---------------------------------------------------------------------------
__global__ void k_edge(const float* __restrict__ x, const int* __restrict__ src,
                       const int* __restrict__ dst, float* __restrict__ dlen,
                       float* __restrict__ du) {
  int e = blockIdx.x * blockDim.x + threadIdx.x;
  if (e >= EE) return;
  int s = src[e], d = dst[e];
  float vx = x[s * 3 + 0] - x[d * 3 + 0];
  float vy = x[s * 3 + 1] - x[d * 3 + 1];
  float vz = x[s * 3 + 2] - x[d * 3 + 2];
  float dd = sqrtf(vx * vx + vy * vy + vz * vz);
  dlen[e] = dd;
  float dn = fmaxf(dd, 1e-8f);
  du[e * 3 + 0] = vx / dn;
  du[e * 3 + 1] = vy / dn;
  du[e * 3 + 2] = vz / dn;
}

// s1[e,j] = bf16( silu(d[e]*em1w[j] + em1b[j]) )
__global__ void k_s1(const float* __restrict__ dlen, const float* __restrict__ w,
                     const float* __restrict__ b, u16* __restrict__ s1) {
  int idx = blockIdx.x * blockDim.x + threadIdx.x;
  if (idx >= ECH * DPP) return;
  int e = idx / DPP, j = idx % DPP;
  float v = dlen[e] * w[j] + b[j];
  v = v / (1.f + __expf(-v));
  s1[idx] = f2bf(v);
}

// mi[e,:] = [ h[src], h[dst], ea ]  -> bf16
__global__ void k_mi(const float* __restrict__ h, const int* __restrict__ src,
                     const int* __restrict__ dst, const float* __restrict__ ea,
                     u16* __restrict__ mi) {
  long idx = (long)blockIdx.x * blockDim.x + threadIdx.x;
  if (idx >= (long)ECH * DIN) return;
  int e = (int)(idx / DIN), j = (int)(idx % DIN);
  float v;
  if (j < CC)            v = h[(long)src[e] * CC + j];
  else if (j < 2 * CC)   v = h[(long)dst[e] * CC + (j - CC)];
  else                   v = ea[(long)e * DPP + (j - 2 * CC)];
  mi[idx] = f2bf(v);
}

__global__ void k_scat_h(const float* __restrict__ m, const int* __restrict__ dst,
                         float* __restrict__ hb) {
  long idx = (long)blockIdx.x * blockDim.x + threadIdx.x;
  if (idx >= (long)ECH * CC) return;
  int e = (int)(idx / CC), j = (int)(idx % CC);
  atomicAdd(&hb[(long)dst[e] * CC + j], m[idx]);
}

__global__ void k_scat_x(const float* __restrict__ cw, const float* __restrict__ du,
                         const int* __restrict__ dst, float* __restrict__ xb) {
  int idx = blockIdx.x * blockDim.x + threadIdx.x;
  if (idx >= ECH * 3) return;
  int e = idx / 3, j = idx % 3;
  atomicAdd(&xb[dst[e] * 3 + j], cw[e * 16] * du[e * 3 + j]);
}

// ---------------------------------------------------------------------------
// frame update (Rodrigues) — updates rot and coords in place
// ---------------------------------------------------------------------------
__global__ void k_frames(float* __restrict__ rot, float* __restrict__ coords,
                         const float* __restrict__ upd) {
  int n = blockIdx.x * blockDim.x + threadIdx.x;
  if (n >= NN) return;
  const float* u = upd + n * 16;
  float rx = u[0], ry = u[1], rz = u[2];
  float tx = u[3], ty = u[4], tz = u[5];
  float ang = fmaxf(sqrtf(rx * rx + ry * ry + rz * rz), 1e-8f);
  float ax = rx / ang, ay = ry / ang, az = rz / ang;
  float ca = cosf(ang), sa = sinf(ang), omc = 1.f - ca;
  float K[9]  = {0.f, -az, ay, az, 0.f, -ax, -ay, ax, 0.f};
  float K2[9];
#pragma unroll
  for (int i = 0; i < 3; ++i)
#pragma unroll
    for (int j = 0; j < 3; ++j) {
      float s = 0.f;
      for (int t = 0; t < 3; ++t) s += K[i * 3 + t] * K[t * 3 + j];
      K2[i * 3 + j] = s;
    }
  float R[9];
#pragma unroll
  for (int i = 0; i < 9; ++i)
    R[i] = ((i % 4) == 0 ? 1.f : 0.f) + sa * K[i] + omc * K2[i];
  float Ro[9];
#pragma unroll
  for (int i = 0; i < 9; ++i) Ro[i] = rot[n * 9 + i];
  float Rn[9];
#pragma unroll
  for (int i = 0; i < 3; ++i)
#pragma unroll
    for (int j = 0; j < 3; ++j) {
      float s = 0.f;
      for (int t = 0; t < 3; ++t) s += Ro[i * 3 + t] * R[t * 3 + j];
      Rn[i * 3 + j] = s;
    }
#pragma unroll
  for (int i = 0; i < 9; ++i) rot[n * 9 + i] = Rn[i];
  float cx = coords[n * 3 + 0], cy = coords[n * 3 + 1], cz = coords[n * 3 + 2];
  coords[n * 3 + 0] = cx + tx * Ro[0] + ty * Ro[3] + tz * Ro[6];
  coords[n * 3 + 1] = cy + tx * Ro[1] + ty * Ro[4] + tz * Ro[7];
  coords[n * 3 + 2] = cz + tx * Ro[2] + ty * Ro[5] + tz * Ro[8];
}

// ---------------------------------------------------------------------------
// Host-side orchestration
// ---------------------------------------------------------------------------
extern "C" void kernel_launch(void* const* d_in, const int* in_sizes, int n_in,
                              void* d_out, int out_size, void* d_ws, size_t ws_size,
                              hipStream_t stream) {
  (void)in_sizes; (void)n_in; (void)out_size; (void)ws_size;
  auto fin = [&](int i) { return (const float*)d_in[i]; };
  // input order: single, pair, frames_rot, frames_trans, mask, then params
  // (ipa: q_w,q_b,k_w,k_b,v_w,v_b,pb_w,pb_b,qp_w,qp_b,kp_w,kp_b,vp_w,vp_b,
  //       out_w,out_b,pout_w,pout_b,ln_g,ln_b; egnn: nm1_w,nm1_b,nm2_w,nm2_b,
  //       cm1_w,cm1_b,cm2_w,em1_w,em1_b,em2_w,em2_b; sm: ru_w,ru_b,t1_w,t1_b,
  //       t2_w,t2_b,ipn_g,ipn_b,n_g,n_b), edge_index
  const float* in_single = fin(0);
  const float* in_pair   = fin(1);
  const float* in_rot    = fin(2);
  const float* in_trans  = fin(3);
  const float* in_mask   = fin(4);
  const int*   edge      = (const int*)d_in[46];
  const int*   src       = edge;
  const int*   dst       = edge + EE;

  // workspace bump allocator (256B aligned)
  char* wp = (char*)d_ws;
  auto alc = [&](size_t bytes) -> char* {
    char* r = wp; wp += (bytes + 255) & ~(size_t)255; return r;
  };
  auto alcF = [&](size_t n) { return (float*)alc(n * sizeof(float)); };
  auto alcH = [&](size_t n) { return (u16*)alc(n * sizeof(u16)); };

  // bf16 transposed weights
  u16* wq    = alcH((size_t)CC * CC);
  u16* wk    = alcH((size_t)CC * CC);
  u16* wv    = alcH((size_t)CC * CC);
  u16* wqp   = alcH((size_t)HP3 * CC);
  u16* wkp   = alcH((size_t)HP3 * CC);
  u16* wvp   = alcH((size_t)HP3 * CC);
  u16* wout  = alcH((size_t)CC * CC);
  u16* wpout = alcH((size_t)CC * HP3P);
  u16* wnm1  = alcH((size_t)HIDD * DIN);
  u16* wnm2  = alcH((size_t)CC * HIDD);
  u16* wcm1  = alcH((size_t)HIDD * DIN);
  u16* wcm2  = alcH((size_t)16 * HIDD);
  u16* wem2  = alcH((size_t)DPP * DPP);
  u16* wru   = alcH((size_t)16 * CC);
  u16* wt1   = alcH((size_t)T4 * CC);
  u16* wt2   = alcH((size_t)CC * T4);

  // activations
  float* singlebuf = alcF((size_t)NN * CC);
  float* rotbuf    = alcF((size_t)NN * 9);
  float* coords    = alcF((size_t)NN * 3);
  u16*   sbf       = alcH((size_t)NN * CC);
  float* qf        = alcF((size_t)NN * CC);
  float* kf        = alcF((size_t)NN * CC);
  float* vf        = alcF((size_t)NN * CC);
  u16*   qbf       = alcH((size_t)NN * CC);
  u16*   kbf       = alcH((size_t)NN * CC);
  u16*   vT        = alcH((size_t)CC * NN);
  float* qp        = alcF((size_t)NN * HP3);
  float* kp        = alcF((size_t)NN * HP3);
  float* vp        = alcF((size_t)NN * HP3);
  float* qg        = alcF((size_t)NN * HP3);
  float* kg        = alcF((size_t)NN * HP3);
  float* vg        = alcF((size_t)NN * HP3);
  u16*   vgT       = alcH((size_t)HP3P * NN);
  float* pb        = alcF((size_t)NN * NN * HH);     // 48 MB
  float* Lb        = alcF((size_t)NN * NN);          // 4 MB
  u16*   attnbf    = alcH((size_t)NN * NN);          // 2 MB
  float* os        = alcF((size_t)NN * CC);
  float* op        = alcF((size_t)NN * HP3);
  u16*   opbf      = alcH((size_t)NN * HP3P);
  u16*   osbf      = alcH((size_t)NN * CC);
  float* aout      = alcF((size_t)NN * CC);
  float* tbig      = alcF((size_t)NN * T4);
  u16*   tbf       = alcH((size_t)NN * T4);
  float* tout      = alcF((size_t)NN * CC);
  float* dlen      = alcF((size_t)EE);
  float* du        = alcF((size_t)EE * 3);
  u16*   s1        = alcH((size_t)ECH * DPP);
  float* ea        = alcF((size_t)ECH * DPP);
  u16*   mibf      = alcH((size_t)ECH * DIN);
  float* m1        = alcF((size_t)ECH * HIDD);
  u16*   m1bf      = alcH((size_t)ECH * HIDD);
  float* mo        = alcF((size_t)ECH * CC);
  float* c1        = alcF((size_t)ECH * HIDD);
  u16*   c1bf      = alcH((size_t)ECH * HIDD);
  float* cw        = alcF((size_t)ECH * 16);
  float* hb        = alcF((size_t)NN * CC);
  float* xb        = alcF((size_t)NN * 3);
  float* upd       = alcF((size_t)NN * 16);

  auto cvtW = [&](const float* W, u16* Bt, int K, int Nc, int Kp, int Np) {
    long n = (long)Np * Kp;
    k_cvtWt<<<(unsigned)((n + 255) / 256), 256, 0, stream>>>(W, Bt, K, Nc, Kp, Np);
  };
  auto cvtA = [&](const float* A, u16* Abf, int M, int K, int Kp) {
    long n = (long)M * Kp;
    k_cvtA<<<(unsigned)((n + 255) / 256), 256, 0, stream>>>(A, Abf, M, K, Kp);
  };
  auto gemm = [&](const u16* A, const u16* Bt, const float* bias, float* C,
                  int M, int Nc, int Kp, int ldc, int act, int accum) {
    int tilesN  = (Nc + 15) / 16;
    int tilesN4 = (tilesN + 3) / 4;
    int tiles = (M / 16) * tilesN4;
    k_gemm<<<(tiles + 7) / 8, 256, 0, stream>>>(A, Bt, bias, C, M, Nc, Kp, ldc,
                                                act, accum);
  };
  auto copyf = [&](const float* a, float* b, long n) {
    k_copy<<<(unsigned)((n + 255) / 256), 256, 0, stream>>>(a, b, n);
  };

  // --- weight prep (bf16, transposed, padded) ---
  cvtW(fin(5),  wq,   CC, CC, CC, CC);
  cvtW(fin(7),  wk,   CC, CC, CC, CC);
  cvtW(fin(9),  wv,   CC, CC, CC, CC);
  cvtW(fin(13), wqp,  CC, HP3, CC, HP3);
  cvtW(fin(15), wkp,  CC, HP3, CC, HP3);
  cvtW(fin(17), wvp,  CC, HP3, CC, HP3);
  cvtW(fin(19), wout, CC, CC, CC, CC);
  cvtW(fin(21), wpout, HP3, CC, HP3P, CC);
  cvtW(fin(25), wnm1, DIN, HIDD, DIN, HIDD);
  cvtW(fin(27), wnm2, HIDD, CC, HIDD, CC);
  cvtW(fin(29), wcm1, DIN, HIDD, DIN, HIDD);
  cvtW(fin(31), wcm2, HIDD, 1, HIDD, 16);
  cvtW(fin(34), wem2, DPP, DPP, DPP, DPP);
  cvtW(fin(36), wru,  CC, 6, CC, 16);
  cvtW(fin(38), wt1,  CC, T4, CC, T4);
  cvtW(fin(40), wt2,  T4, CC, T4, CC);

  // --- mutable state init ---
  copyf(in_single, singlebuf, (long)NN * CC);
  copyf(in_rot,    rotbuf,    (long)NN * 9);
  copyf(in_trans,  coords,    (long)NN * 3);

  for (int blk = 0; blk < 2; ++blk) {
    // ======================= IPA =======================
    cvtA(singlebuf, sbf, NN, CC, CC);
    gemm(sbf, wq, fin(6),  qf, NN, CC, CC, CC, 0, 0);
    gemm(sbf, wk, fin(8),  kf, NN, CC, CC, CC, 0, 0);
    gemm(sbf, wv, fin(10), vf, NN, CC, CC, CC, 0, 0);
    gemm(sbf, wqp, fin(14), qp, NN, HP3, CC, HP3, 0, 0);
    gemm(sbf, wkp, fin(16), kp, NN, HP3, CC, HP3, 0, 0);
    gemm(sbf, wvp, fin(18), vp, NN, HP3, CC, HP3, 0, 0);
    {
      int n = NN * HH * PP;
      k_glob<<<(n + 255) / 256, 256, 0, stream>>>(qp, rotbuf, coords, qg);
      k_glob<<<(n + 255) / 256, 256, 0, stream>>>(kp, rotbuf, coords, kg);
      k_glob<<<(n + 255) / 256, 256, 0, stream>>>(vp, rotbuf, coords, vg);
    }
    cvtA(qf, qbf, NN, CC, CC);
    cvtA(kf, kbf, NN, CC, CC);
    cvtW(vf, vT,  NN, CC, NN, CC);       // v columns -> Bt[384,1024]
    cvtW(vg, vgT, NN, HP3, NN, HP3P);    // vg columns -> Bt[160,1024]
    {
      long waves = (long)NN * NN;
      k_pair_bias<<<(unsigned)(waves / 8), 256, 0, stream>>>(in_pair, fin(11),
                                                             fin(12), pb);
    }
    for (int h = 0; h < HH; ++h) {
      int tiles = (NN / 16) * (NN / 16);
      k_attn_logits<<<tiles / 8, 256, 0, stream>>>(qbf, kbf, h * CHH, qg, kg,
                                                   h * 12, pb, h, in_mask, Lb);
      k_softmax<<<NN, 256, 0, stream>>>(Lb, attnbf);
      gemm(attnbf, vT  + (size_t)h * CHH * NN, nullptr, os + h * CHH,
           NN, CHH, NN, CC, 0, 0);
      gemm(attnbf, vgT + (size_t)h * 12 * NN, nullptr, op + h * 12,
           NN, 12, NN, HP3, 0, 0);
    }
    cvtA(op, opbf, NN, HP3, HP3P);
    gemm(opbf, wpout, fin(22), os, NN, CC, HP3P, CC, 0, 1);   // os += op@pout + b
    cvtA(os, osbf, NN, CC, CC);
    gemm(osbf, wout, fin(20), aout, NN, CC, CC, CC, 0, 0);
    k_add_ln<<<NN, 128, 0, stream>>>(aout, singlebuf, fin(23), fin(24), singlebuf);
    k_add_ln<<<NN, 128, 0, stream>>>(singlebuf, nullptr, fin(42), fin(43), singlebuf);

    // =================== transition ====================
    cvtA(singlebuf, sbf, NN, CC, CC);
    gemm(sbf, wt1, fin(39), tbig, NN, T4, CC, T4, 1, 0);       // relu
    cvtA(tbig, tbf, NN, T4, T4);
    gemm(tbf, wt2, fin(41), tout, NN, CC, T4, CC, 0, 0);
    k_add_ln<<<NN, 128, 0, stream>>>(tout, singlebuf, fin(44), fin(45), singlebuf);

    // ====================== EGNN =======================
    k_edge<<<(EE + 255) / 256, 256, 0, stream>>>(coords, src, dst, dlen, du);
    copyf(singlebuf, hb, (long)NN * CC);
    copyf(coords, xb, (long)NN * 3);
    for (int c = 0; c < EE / ECH; ++c) {
      int e0 = c * ECH;
      k_s1<<<(ECH * DPP + 255) / 256, 256, 0, stream>>>(dlen + e0, fin(32),
                                                        fin(33), s1);
      gemm(s1, wem2, fin(35), ea, ECH, DPP, DPP, DPP, 0, 0);
      {
        long n = (long)ECH * DIN;
        k_mi<<<(unsigned)((n + 255) / 256), 256, 0, stream>>>(singlebuf, src + e0,
                                                              dst + e0, ea, mibf);
      }
      gemm(mibf, wnm1, fin(26), m1, ECH, HIDD, DIN, HIDD, 2, 0);   // silu
      cvtA(m1, m1bf, ECH, HIDD, HIDD);
      gemm(m1bf, wnm2, fin(28), mo, ECH, CC, HIDD, CC, 0, 0);
      {
        long n = (long)ECH * CC;
        k_scat_h<<<(unsigned)((n + 255) / 256), 256, 0, stream>>>(mo, dst + e0, hb);
      }
      gemm(mibf, wcm1, fin(30), c1, ECH, HIDD, DIN, HIDD, 2, 0);   // silu
      cvtA(c1, c1bf, ECH, HIDD, HIDD);
      gemm(c1bf, wcm2, nullptr, cw, ECH, 1, HIDD, 16, 0, 0);
      k_scat_x<<<(ECH * 3 + 255) / 256, 256, 0, stream>>>(cw, du + (size_t)e0 * 3,
                                                          dst + e0, xb);
    }
    copyf(hb, singlebuf, (long)NN * CC);
    copyf(xb, coords, (long)NN * 3);

    // =================== frame update ==================
    cvtA(singlebuf, sbf, NN, CC, CC);
    gemm(sbf, wru, fin(37), upd, NN, 6, CC, 16, 0, 0);
    k_frames<<<(NN + 255) / 256, 256, 0, stream>>>(rotbuf, coords, upd);
  }

  // outputs: single [1,1024,384] then coords [1,1024,3]
  copyf(singlebuf, (float*)d_out, (long)NN * CC);
  copyf(coords, (float*)d_out + (long)NN * CC, (long)NN * 3);
}
